// FeatureCrossAttention_36876589203611
// MI455X (gfx1250) — compile-verified, matrix operand order checked by reading, not run on hardware
//
#include <hip/hip_runtime.h>
#include <hip/hip_bf16.h>

// FeatureCrossAttention, MI455X (gfx1250, wave32, WMMA).
//
// Structure exploited:
//   Q_i rank-1, K_i/V_i rank-1 updates of K0/V0, output projection hoisted.
//   Heavy math = 16x per-(b,h) attention (QK^T + P@V) in bf16 WMMA, f32 acc.

typedef __attribute__((ext_vector_type(16))) __bf16 v16bf;
typedef __attribute__((ext_vector_type(8)))  float  v8f;

#define S_LEN   512
#define BATCH   16
#define DMODEL  512
#define NHEAD   8
#define DHEAD   64
#define NITER   16
#define QK_SCALE 0.125f   // DHEAD^-0.5

static __device__ __forceinline__ v8f wmma_bf16(v16bf a, v16bf b, v8f c) {
  // v_wmma_f32_16x16x32_bf16: D = A(16x32) * B(32x16) + C(16x16 f32)
  return __builtin_amdgcn_wmma_f32_16x16x32_bf16(false, a, false, b, (short)0, c,
                                                 false, false);
}

// ---------------------------------------------------------------- cvt f32->bf16
__global__ __launch_bounds__(256)
void cvt_f32_bf16(const float* __restrict__ in, __bf16* __restrict__ out, int n) {
  int i = blockIdx.x * 256 + threadIdx.x;
  int stride = gridDim.x * 256;
  for (; i < n; i += stride) out[i] = (__bf16)in[i];
}

// ------------------------------------------------- C[MxN] = A[MxK] @ B[NxK]^T
// + biasScale*bias[N].  A, B bf16 row-major K-contiguous; C f32.
// Block = 256 threads (8 waves); wave computes a 16x64 tile; WG = 128x64.
__global__ __launch_bounds__(256)
void gemm_ntn_bias(const __bf16* __restrict__ A, const __bf16* __restrict__ Bw,
                   const float* __restrict__ bias, float biasScale,
                   float* __restrict__ C, int M, int N, int K) {
  const int lane = threadIdx.x & 31;
  const int wave = threadIdx.x >> 5;
  const int nblk = N / 64;
  const int bm = blockIdx.x / nblk;
  const int bn = blockIdx.x % nblk;
  const int m0 = bm * 128 + wave * 16;
  const int n0 = bn * 64;
  const int lm = lane & 15;
  const int koff = (lane >> 4) * 16;   // A/B k order chosen here (contiguous): hi-half holds K+16
  const int mrow = m0 + lm;

  v8f acc0 = {}, acc1 = {}, acc2 = {}, acc3 = {};
  for (int k = 0; k < K; k += 32) {
    v16bf a  = *(const v16bf*)(A  + (size_t)mrow * K + k + koff);
    v16bf b0 = *(const v16bf*)(Bw + (size_t)(n0 +  0 + lm) * K + k + koff);
    v16bf b1 = *(const v16bf*)(Bw + (size_t)(n0 + 16 + lm) * K + k + koff);
    v16bf b2 = *(const v16bf*)(Bw + (size_t)(n0 + 32 + lm) * K + k + koff);
    v16bf b3 = *(const v16bf*)(Bw + (size_t)(n0 + 48 + lm) * K + k + koff);
    acc0 = wmma_bf16(a, b0, acc0);
    acc1 = wmma_bf16(a, b1, acc1);
    acc2 = wmma_bf16(a, b2, acc2);
    acc3 = wmma_bf16(a, b3, acc3);
  }
  const int rbase = m0 + 8 * (lane >> 4);
#define GEMM_STORE(nt, accv)                                                   \
  {                                                                            \
    int ncol = n0 + nt * 16 + lm;                                              \
    float bb = biasScale * bias[ncol];                                         \
    _Pragma("unroll") for (int r = 0; r < 8; ++r)                              \
        C[(size_t)(rbase + r) * N + ncol] = accv[r] + bb;                      \
  }
  GEMM_STORE(0, acc0) GEMM_STORE(1, acc1) GEMM_STORE(2, acc2) GEMM_STORE(3, acc3)
#undef GEMM_STORE
}

// ------------------------------------------------------------- fused attention
// grid.x = (b,h) * 4 M-blocks = 512;  block = 256 (8 waves, 16 query rows each).
// Loops i=0..15 inside so O accumulates in registers (no atomics).
__global__ __launch_bounds__(256)
void attn_kernel(const float* __restrict__ x, const float* __restrict__ Win,
                 const float* __restrict__ bin,
                 const float* __restrict__ K0, const float* __restrict__ V0,
                 __bf16* __restrict__ Obf) {
  __shared__ __align__(32) __bf16 Klds[64][80];      // [t_local][dh]
  __shared__ __align__(32) __bf16 VldsT[64][80];     // [dh][t_local] (transposed)
  __shared__ __align__(32) __bf16 Plds[8][16][80];   // per-wave P tile [m][t_local]
  __shared__ float xcol[S_LEN];
  __shared__ float wqc[DHEAD], wkc[DHEAD], wvc[DHEAD], bqv[DHEAD];

  const int tid  = threadIdx.x;
  const int lane = tid & 31;
  const int wave = tid >> 5;
  const int mb = blockIdx.x & 3;
  const int bh = blockIdx.x >> 2;
  const int b  = bh >> 3;
  const int h  = bh & 7;
  const int s0 = mb * 128 + wave * 16;
  const int lm = lane & 15;
  const int lh = lane >> 4;

  v8f accO0 = {}, accO1 = {}, accO2 = {}, accO3 = {};

  for (int i = 0; i < NITER; ++i) {
    // ---- stage per-(b,i) channel column and weight columns
    for (int idx = tid; idx < S_LEN; idx += 256)
      xcol[idx] = x[((size_t)idx * BATCH + b) * DMODEL + i];
    for (int idx = tid; idx < DHEAD; idx += 256) {
      int row = h * DHEAD + idx;
      wqc[idx] = Win[(size_t)row * DMODEL + i];
      wkc[idx] = Win[(size_t)(DMODEL + row) * DMODEL + i];
      wvc[idx] = Win[(size_t)(2 * DMODEL + row) * DMODEL + i];
      bqv[idx] = bin[row];
    }
    __syncthreads();

    // ---- build rank-1 Q as WMMA A-operands (2 K=32 slabs)
    v16bf qa0, qa1;
    {
      float xs = xcol[s0 + lm];
#pragma unroll
      for (int j = 0; j < 16; ++j) {
        int dh0 = lh * 16 + j;
        int dh1 = 32 + lh * 16 + j;
        qa0[j] = (__bf16)((xs * wqc[dh0] + bqv[dh0]) * QK_SCALE);
        qa1[j] = (__bf16)((xs * wqc[dh1] + bqv[dh1]) * QK_SCALE);
      }
    }

    float m_r[8], l_r[8];
#pragma unroll
    for (int r = 0; r < 8; ++r) { m_r[r] = -3.0e38f; l_r[r] = 0.0f; }
    v8f O0 = {}, O1 = {}, O2 = {}, O3 = {};

    for (int tc = 0; tc < S_LEN / 64; ++tc) {
      __syncthreads();  // previous chunk fully consumed
      // ---- stage K/V chunk with rank-1 update applied; V transposed
      for (int e = tid; e < 64 * 64; e += 256) {
        int tl = e >> 6, dh = e & 63;
        int t = tc * 64 + tl;
        size_t g = ((size_t)t * BATCH + b) * DMODEL + h * DHEAD + dh;
        float xv = xcol[t];
        Klds[tl][dh]  = (__bf16)(K0[g] - xv * wkc[dh]);
        VldsT[dh][tl] = (__bf16)(V0[g] - xv * wvc[dh]);
        if (tc + 1 < S_LEN / 64) {
          __builtin_prefetch(K0 + g + (size_t)64 * BATCH * DMODEL, 0, 1);
          __builtin_prefetch(V0 + g + (size_t)64 * BATCH * DMODEL, 0, 1);
        }
      }
      __syncthreads();

      // ---- scores S_tile(16x64): 4 N-subtiles, each 2 chained WMMAs over dh
      v8f st[4];
#pragma unroll
      for (int nt = 0; nt < 4; ++nt) {
        int tl = nt * 16 + lm;
        v16bf kb0 = *(const v16bf*)&Klds[tl][lh * 16];
        v16bf kb1 = *(const v16bf*)&Klds[tl][32 + lh * 16];
        v8f z = {};
        z = wmma_bf16(qa0, kb0, z);
        z = wmma_bf16(qa1, kb1, z);
        st[nt] = z;
      }

      // ---- online softmax (row = VGPR slot + half; reduce across 16-lane half)
#pragma unroll
      for (int r = 0; r < 8; ++r) {
        float v = fmaxf(fmaxf(st[0][r], st[1][r]), fmaxf(st[2][r], st[3][r]));
#pragma unroll
        for (int off = 8; off >= 1; off >>= 1)
          v = fmaxf(v, __shfl_xor(v, off, 16));
        float mnew = fmaxf(m_r[r], v);
        float sc = __expf(m_r[r] - mnew);
        m_r[r] = mnew;
        l_r[r] *= sc;
        O0[r] *= sc; O1[r] *= sc; O2[r] *= sc; O3[r] *= sc;
        float ps = 0.0f;
#pragma unroll
        for (int nt = 0; nt < 4; ++nt) {
          float p = __expf(st[nt][r] - mnew);
          st[nt][r] = p;
          ps += p;
        }
#pragma unroll
        for (int off = 8; off >= 1; off >>= 1)
          ps += __shfl_xor(ps, off, 16);
        l_r[r] += ps;
      }

      // ---- P (C-layout) -> LDS bf16 (row-major) for A-operand reload
#pragma unroll
      for (int nt = 0; nt < 4; ++nt)
#pragma unroll
        for (int r = 0; r < 8; ++r)
          Plds[wave][r + 8 * lh][nt * 16 + lm] = (__bf16)st[nt][r];
      // wave-private region; LDS ops in-order within a wave -> no barrier needed

      // ---- O += P(16x64) @ V(64x64)
      v16bf pa0 = *(const v16bf*)&Plds[wave][lm][lh * 16];
      v16bf pa1 = *(const v16bf*)&Plds[wave][lm][32 + lh * 16];
#define PV_STEP(dt, Ov)                                                        \
  {                                                                            \
    v16bf vb0 = *(const v16bf*)&VldsT[dt * 16 + lm][lh * 16];                  \
    v16bf vb1 = *(const v16bf*)&VldsT[dt * 16 + lm][32 + lh * 16];             \
    Ov = wmma_bf16(pa0, vb0, Ov);                                              \
    Ov = wmma_bf16(pa1, vb1, Ov);                                              \
  }
      PV_STEP(0, O0) PV_STEP(1, O1) PV_STEP(2, O2) PV_STEP(3, O3)
#undef PV_STEP
    }

    // ---- normalize this iteration's output and accumulate
    float inv[8];
#pragma unroll
    for (int r = 0; r < 8; ++r) inv[r] = 1.0f / l_r[r];
#pragma unroll
    for (int r = 0; r < 8; ++r) {
      accO0[r] += O0[r] * inv[r];
      accO1[r] += O1[r] * inv[r];
      accO2[r] += O2[r] * inv[r];
      accO3[r] += O3[r] * inv[r];
    }
    __syncthreads();
  }

  // ---- store accumulated O as bf16 in (S,B,H*DH) layout for final projection
#define O_STORE(dt, accv)                                                      \
  {                                                                            \
    int col = h * DHEAD + dt * 16 + lm;                                        \
    _Pragma("unroll") for (int r = 0; r < 8; ++r) {                            \
      int s = s0 + r + 8 * lh;                                                 \
      Obf[((size_t)s * BATCH + b) * DMODEL + col] = (__bf16)accv[r];           \
    }                                                                          \
  }
  O_STORE(0, accO0) O_STORE(1, accO1) O_STORE(2, accO2) O_STORE(3, accO3)
#undef O_STORE
}

// --------------------------------------------------------------------- launch
extern "C" void kernel_launch(void* const* d_in, const int* in_sizes, int n_in,
                              void* d_out, int out_size, void* d_ws, size_t ws_size,
                              hipStream_t stream) {
  const float* x    = (const float*)d_in[0];   // (512,16,512)
  const float* Win  = (const float*)d_in[1];   // (1536,512) packed Wq|Wk|Wv
  const float* bin  = (const float*)d_in[2];   // (1536,)
  const float* Wout = (const float*)d_in[3];   // (512,512)
  const float* bout = (const float*)d_in[4];   // (512,)
  float* out = (float*)d_out;                  // (512,16,512) f32

  const int M = S_LEN * BATCH;                 // 8192 flattened (s,b) rows
  char* ws = (char*)d_ws;
  __bf16* xbf  = (__bf16*)(ws);                // 8 MB
  __bf16* wkbf = (__bf16*)(ws +  8388608);     // 512 KB
  __bf16* wvbf = (__bf16*)(ws +  8912896);     // 512 KB
  __bf16* wobf = (__bf16*)(ws +  9437184);     // 512 KB
  float*  K0   = (float*) (ws +  9961472);     // 16 MB
  float*  V0   = (float*) (ws + 26738688);     // 16 MB
  __bf16* Obf  = (__bf16*)(ws + 43515904);     // 8 MB  (total ~51.9 MB)

  // bf16 copies for WMMA GEMMs
  cvt_f32_bf16<<<512, 256, 0, stream>>>(x, xbf, M * DMODEL);
  cvt_f32_bf16<<<128, 256, 0, stream>>>(Win + DMODEL * DMODEL, wkbf, DMODEL * DMODEL);
  cvt_f32_bf16<<<128, 256, 0, stream>>>(Win + 2 * DMODEL * DMODEL, wvbf, DMODEL * DMODEL);
  cvt_f32_bf16<<<128, 256, 0, stream>>>(Wout, wobf, DMODEL * DMODEL);

  // K0 = x @ Wk^T + bk ; V0 = x @ Wv^T + bv  (computed once, reused 16x)
  const int gemmBlocks = (M / 128) * (DMODEL / 64);  // 512
  gemm_ntn_bias<<<gemmBlocks, 256, 0, stream>>>(xbf, wkbf, bin + DMODEL, 1.0f,
                                                K0, M, DMODEL, DMODEL);
  gemm_ntn_bias<<<gemmBlocks, 256, 0, stream>>>(xbf, wvbf, bin + 2 * DMODEL, 1.0f,
                                                V0, M, DMODEL, DMODEL);

  // 16 fused attention passes, O accumulated in registers
  attn_kernel<<<512, 256, 0, stream>>>(x, Win, bin, K0, V0, Obf);

  // out = (sum_i o_i) @ Wout^T + 16 * b_out
  gemm_ntn_bias<<<gemmBlocks, 256, 0, stream>>>(Obf, wobf, bout, 16.0f,
                                                out, M, DMODEL, DMODEL);
}
